// GatedGN_214748365021
// MI455X (gfx1250) — matrix operand order repaired and checked by reading, not compiled, not run on hardware
//
#include <hip/hip_runtime.h>
#include <math.h>

// ---------------------------------------------------------------------------
// CDNA5 (gfx1250) WMMA bf16 implementation of the GatedGN reference.
// N=4096, K=8, n_in=14, H=128, k=3. All GEMMs via v_wmma_f32_16x16x32_bf16.
// Each wave owns one 16-wide output-column block and accumulates 4 row-tiles
// (64 rows/block) so B-fragments are loaded once per k-chunk and reused.
// Tile staging uses GLOBAL_LOAD_ASYNC_TO_LDS_B128 (ASYNCcnt) for the bf16
// spans, overlapped with the f32->bf16 state conversion.
// ---------------------------------------------------------------------------

typedef __attribute__((ext_vector_type(16))) __bf16 v16bf;
typedef __attribute__((ext_vector_type(8)))  float  v8f;

#define WGSZ   256          // 8 waves of 32
#define GRROWS 64           // rows per block in GRU kernels (4 WMMA M-tiles)
#define EROWS  64           // rows per block in encoder kernels

__device__ __forceinline__ __bf16 f2bf(float x) { return (__bf16)x; }

__device__ __forceinline__ float tanh_fast(float x) {
#if __has_builtin(__builtin_amdgcn_tanhf)
  return __builtin_amdgcn_tanhf(x);          // v_tanh_f32 (gfx1250)
#elif __has_builtin(__builtin_amdgcn_tanh_f32)
  return __builtin_amdgcn_tanh_f32(x);
#else
  return tanhf(x);
#endif
}
__device__ __forceinline__ float rcp_fast(float x) {
#if __has_builtin(__builtin_amdgcn_rcpf)
  return __builtin_amdgcn_rcpf(x);           // v_rcp_f32
#else
  return 1.0f / x;
#endif
}
__device__ __forceinline__ float sigf(float x) { return rcp_fast(1.0f + __expf(-x)); }

// Async copy of one 16-byte chunk global -> LDS (GLOBAL_LOAD_ASYNC_TO_LDS_B128,
// tracked on ASYNCcnt). lds_off is the LDS byte address (addr[31:0] of the
// generic pointer), gaddr the 64-bit global byte address; both 16B-aligned.
__device__ __forceinline__ void async_g2l_b128(unsigned lds_off,
                                               unsigned long long gaddr) {
  asm volatile("global_load_async_to_lds_b128 %0, %1, off"
               :: "v"(lds_off), "v"(gaddr) : "memory");
}
__device__ __forceinline__ void wait_async0() {
  asm volatile("s_wait_asynccnt 0x0" ::: "memory");
}
__device__ __forceinline__ unsigned lds_off_of(const void* p) {
  return (unsigned)(unsigned long long)(size_t)p;   // low 32 bits = LDS offset
}

// ---- WMMA fragment helpers (wave32; A and B share contiguous k = hi*16+t) ----------

// A: 16x32 bf16 tile, row-major, stride ld. Per lane: M=lane&15, hi=lane>>4;
// 16 contiguous k: k = hi*16+t. Two 16B LDS reads.
__device__ __forceinline__ v16bf load_a_frag(const __bf16* base, int ld) {
  const int lane = threadIdx.x & 31;
  const int m = lane & 15, hi = lane >> 4;
  const __bf16* r0 = base + m * ld + hi * 16;
  v16bf a;
#pragma unroll
  for (int t = 0; t < 8; ++t)  a[t] = r0[t];
#pragma unroll
  for (int t = 8; t < 16; ++t) a[t] = r0[t];
  return a;
}

// B: W row-major (cols x Ktot) == pre-transposed for X @ W^T. Per lane:
// N = c0+(lane&15); 16 contiguous k at k0 + (lane>>4)*16. Two 16B global reads.
__device__ __forceinline__ v16bf load_b_frag(const __bf16* W, int ld, int c0, int k0) {
  const int lane = threadIdx.x & 31;
  const int n = lane & 15, hi = lane >> 4;
  const __bf16* row = W + (size_t)(c0 + n) * ld + k0 + hi * 16;
  v16bf b;
#pragma unroll
  for (int t = 0; t < 16; ++t) b[t] = row[t];
  return b;
}

__device__ __forceinline__ v8f wmma_bf16(v16bf a, v16bf b, v8f c) {
  return __builtin_amdgcn_wmma_f32_16x16x32_bf16(false, a, false, b, (short)0, c,
                                                 false, false);
}

// ---------------------------------------------------------------------------
// Weight / bias packing kernels (tiny, once per launch)
// ---------------------------------------------------------------------------

__global__ void pack_pad_bf16(__bf16* __restrict__ dst, const float* __restrict__ src,
                              int rows, int scols, int dcols) {
  int tid = blockIdx.x * blockDim.x + threadIdx.x;
  if (tid >= rows * dcols) return;
  int r = tid / dcols, c = tid - r * dcols;
  float v = (c < scols) ? src[r * scols + c] : 0.0f;
  dst[tid] = f2bf(v);
}

// Wcat (512 x (XW+128)): rows [0,256)=r,z over [Wih|Whh]; [256,384)=i_n (Wih);
// [384,512)=h_n (Whh). Each 142-wide Wih x-half padded to 160.
__global__ void pack_gru_cat(__bf16* __restrict__ dst, const float* __restrict__ Wih,
                             const float* __restrict__ Whh, int XW, int ihw) {
  const int TW = XW + 128;
  int tid = blockIdx.x * blockDim.x + threadIdx.x;
  if (tid >= 512 * TW) return;
  int r = tid / TW, c = tid - r * TW;
  float v = 0.0f;
  if (c < XW) {
    if (r < 384) {
      int half = c / 160, cc = c - half * 160;
      if (cc < 142) v = Wih[r * ihw + half * 142 + cc];
    }
  } else {
    if (r < 256) v = Whh[r * 128 + (c - XW)];
    else if (r >= 384) v = Whh[(r - 128) * 128 + (c - XW)];
  }
  dst[tid] = f2bf(v);
}

// bias[512] = [bih+bhh (r,z) | bih_n | bhh_n]
__global__ void pack_bias(float* __restrict__ dst, const float* __restrict__ bih,
                          const float* __restrict__ bhh) {
  int c = blockIdx.x * blockDim.x + threadIdx.x;
  if (c >= 512) return;
  float v;
  if (c < 256) v = bih[c] + bhh[c];
  else if (c < 384) v = bih[c];
  else v = bhh[c - 128];
  dst[c] = v;
}

// ---------------------------------------------------------------------------
// Fused two-layer tanh MLP encoder (mode 0: nodes E_n, mode 1: edges E_e)
// 64 rows/block; wave ct owns one 16-col block, accumulates 4 row-tiles.
// ---------------------------------------------------------------------------
__global__ __launch_bounds__(WGSZ)
void encoder_mlp(const float* __restrict__ towers, const __bf16* __restrict__ W1,
                 const float* __restrict__ b1, const __bf16* __restrict__ W2,
                 const float* __restrict__ b2, float* __restrict__ out,
                 long rows, int mode) {
  extern __shared__ char smem[];
  __bf16* X0 = (__bf16*)smem;                       // 64 x 32
  __bf16* T1 = (__bf16*)(smem + EROWS * 32 * 2);    // 64 x 128
  const long rowBase = (long)blockIdx.x * EROWS;

  // Phase A: gather input tile (padded bf16)
  for (int idx = threadIdx.x; idx < EROWS * 32; idx += WGSZ) {
    int r = idx >> 5, c = idx & 31;
    long g = rowBase + r;
    float v = 0.0f;
    if (g < rows) {
      if (mode == 0) {
        if (c < 14) v = towers[g * 14 + c];
      } else {
        long j = g & 7, ni = g >> 3, n = ni >> 3;      // row = (n*8+i)*8+j
        if (c < 14)      v = towers[ni * 14 + c];
        else if (c < 28) v = towers[(n * 8 + j) * 14 + (c - 14)];
      }
    }
    X0[idx] = f2bf(v);
  }
  __syncthreads();

  const int ct = threadIdx.x >> 5;                    // wave id == col tile
  const int lane = threadIdx.x & 31, n = lane & 15, hi = lane >> 4;

  // Layer 1: K = 32 (1 chunk); B loaded once, reused across 4 row-tiles
  {
    v16bf b = load_b_frag(W1, 32, ct * 16, 0);
    v8f acc[4] = {};
#pragma unroll
    for (int rt = 0; rt < 4; ++rt) {
      v16bf a = load_a_frag(X0 + rt * 16 * 32, 32);
      acc[rt] = wmma_bf16(a, b, acc[rt]);
    }
    const float b1c = b1[ct * 16 + n];
#pragma unroll
    for (int rt = 0; rt < 4; ++rt)
#pragma unroll
      for (int v = 0; v < 8; ++v)
        T1[(rt * 16 + v + hi * 8) * 128 + ct * 16 + n] =
            f2bf(tanh_fast(acc[rt][v] + b1c));
  }
  __syncthreads();

  // Layer 2: K = 128 (4 chunks)
  {
    v8f acc[4] = {};
#pragma unroll
    for (int kc = 0; kc < 4; ++kc) {
      v16bf b = load_b_frag(W2, 128, ct * 16, kc * 32);
#pragma unroll
      for (int rt = 0; rt < 4; ++rt) {
        v16bf a = load_a_frag(T1 + rt * 16 * 128 + kc * 32, 128);
        acc[rt] = wmma_bf16(a, b, acc[rt]);
      }
    }
    const float b2c = b2[ct * 16 + n];
#pragma unroll
    for (int rt = 0; rt < 4; ++rt)
#pragma unroll
      for (int v = 0; v < 8; ++v) {
        long g = rowBase + rt * 16 + v + hi * 8;
        if (g < rows) out[g * 128 + ct * 16 + n] = tanh_fast(acc[rt][v] + b2c);
      }
  }
}

// ---------------------------------------------------------------------------
// Input staging kernels
// ---------------------------------------------------------------------------

// xh[(n,i), 0:160] = [towers(14) | h(128) | pad]  (bf16)
__global__ void build_xh(__bf16* __restrict__ xh, const float* __restrict__ towers,
                         const float* __restrict__ h, long nk) {
  long tid = (long)blockIdx.x * blockDim.x + threadIdx.x;
  if (tid >= nk * 160) return;
  long r = tid / 160; int c = (int)(tid - r * 160);
  float v = 0.0f;
  if (c < 14)       v = towers[r * 14 + c];
  else if (c < 142) v = h[r * 128 + (c - 14)];
  xh[tid] = f2bf(v);
}

// nx[(n,i), 0:160] = [towers(14) | e[n,i,i+1](128, 0 for i==7) | pad]
__global__ void build_nx(__bf16* __restrict__ nx, const float* __restrict__ towers,
                         const float* __restrict__ e, long nk) {
  long tid = (long)blockIdx.x * blockDim.x + threadIdx.x;
  if (tid >= nk * 160) return;
  long r = tid / 160; int c = (int)(tid - r * 160);
  float v = 0.0f;
  if (c < 14) v = towers[r * 14 + c];
  else if (c < 142) {
    int i = (int)(r & 7);
    if (i < 7) v = e[(r * 8 + i + 1) * 128 + (c - 14)];
  }
  nx[tid] = f2bf(v);
}

// ---------------------------------------------------------------------------
// GRU cell over packed Wcat (512 x TW). 64 rows/block. Each wave owns one
// 16-wide H-column block hc and computes ALL FOUR gate tiles for 4 row-tiles;
// gates are evaluated directly from accumulator fragments (identical C/D
// layout across the four WMMAs) -> no Out LDS, single barrier.
// Phase A: bf16 x-spans staged with async global->LDS B128 copies (ASYNCcnt),
// overlapped with the f32->bf16 recurrent-state conversion.
//   mode 0 (edge): row=(n,i,j): X = [xh[n,i] | xh[n,j] | e(row)]   TW=448
//   mode 1 (node): row=(n,i):   X = [nx[row] | h[row]]             TW=288
// Grids divide rows exactly (rows % 64 == 0), so no row guard in Phase A.
// ---------------------------------------------------------------------------
__global__ __launch_bounds__(WGSZ)
void gru_cell(const __bf16* __restrict__ Xsrc, float* __restrict__ state,
              const __bf16* __restrict__ Wcat, const float* __restrict__ bias,
              long rows, int mode) {
  const int XW = (mode == 0) ? 320 : 160;
  const int TW = XW + 128;
  const int KX = XW >> 5;          // k-chunks covering x part
  const int KT = TW >> 5;          // total k-chunks
  extern __shared__ char smem[];
  __bf16* X = (__bf16*)smem;       // GRROWS x TW bf16
  const long rowBase = (long)blockIdx.x * GRROWS;
  const int lane = threadIdx.x & 31;
  const int wv = threadIdx.x >> 5;

  // Warm L2 with the packed weight rows this block is about to stream.
  __builtin_prefetch(Wcat + (size_t)(threadIdx.x * 2) * TW, 0, 2);

  // Phase A: per-wave row staging. Each 320B bf16 x-span moves via 20 lanes of
  // async B128 copies; the 128-wide f32 state converts to bf16 via VALU.
  for (int r = wv; r < GRROWS; r += 8) {
    const long g = rowBase + r;
    const unsigned lbase = lds_off_of(X + r * TW);
    if (mode == 0) {
      const long j = g & 7, ni = g >> 3, n = ni >> 3;
      if (lane < 20) {
        async_g2l_b128(lbase + lane * 16,
                       (unsigned long long)(size_t)(Xsrc + ni * 160) + lane * 16);
        async_g2l_b128(lbase + 320 + lane * 16,
                       (unsigned long long)(size_t)(Xsrc + (n * 8 + j) * 160) + lane * 16);
      }
    } else {
      if (lane < 20) {
        async_g2l_b128(lbase + lane * 16,
                       (unsigned long long)(size_t)(Xsrc + g * 160) + lane * 16);
      }
    }
    const float* sp = state + g * 128;
    __bf16* dp = X + r * TW + XW;
    for (int c = lane; c < 128; c += 32) dp[c] = f2bf(sp[c]);
  }
  wait_async0();                 // drain ASYNCcnt (inline-asm ops are untracked)
  __syncthreads();

  // Phase B: fused 4-gate GEMM. Wave-uniform control flow; EXEC stays full.
  const int hc = wv;                                  // 0..7: H column block
  v8f aR[4] = {}, aZ[4] = {}, aN[4] = {}, aH[4] = {};
  for (int kc = 0; kc < KT; ++kc) {
    v16bf bR = load_b_frag(Wcat, TW, hc * 16,       kc * 32);
    v16bf bZ = load_b_frag(Wcat, TW, 128 + hc * 16, kc * 32);
    if (kc < KX) {
      v16bf bN = load_b_frag(Wcat, TW, 256 + hc * 16, kc * 32);
#pragma unroll
      for (int rt = 0; rt < 4; ++rt) {
        v16bf a = load_a_frag(X + rt * 16 * TW + kc * 32, TW);
        aR[rt] = wmma_bf16(a, bR, aR[rt]);
        aZ[rt] = wmma_bf16(a, bZ, aZ[rt]);
        aN[rt] = wmma_bf16(a, bN, aN[rt]);
      }
    } else {
      v16bf bH = load_b_frag(Wcat, TW, 384 + hc * 16, kc * 32);
#pragma unroll
      for (int rt = 0; rt < 4; ++rt) {
        v16bf a = load_a_frag(X + rt * 16 * TW + kc * 32, TW);
        aR[rt] = wmma_bf16(a, bR, aR[rt]);
        aZ[rt] = wmma_bf16(a, bZ, aZ[rt]);
        aH[rt] = wmma_bf16(a, bH, aH[rt]);
      }
    }
  }

  // Phase C: gates + in-place state update straight from fragments.
  const int n = lane & 15, hi = lane >> 4;
  const int c = hc * 16 + n;
  const float bRc = bias[c], bZc = bias[128 + c];
  const float bNc = bias[256 + c], bHc = bias[384 + c];
#pragma unroll
  for (int rt = 0; rt < 4; ++rt) {
#pragma unroll
    for (int v = 0; v < 8; ++v) {
      long g = rowBase + rt * 16 + v + hi * 8;
      if (g < rows) {
        float rg = sigf(aR[rt][v] + bRc);
        float zg = sigf(aZ[rt][v] + bZc);
        float ng = tanh_fast(fmaf(rg, aH[rt][v] + bHc, aN[rt][v] + bNc));
        float hp = state[g * 128 + c];
        state[g * 128 + c] = fmaf(zg, hp - ng, ng);   // (1-z)n + z h
      }
    }
  }
}

// gx[n, 0:256] = [sum_i h[n,i] | sum_{i<7} e[n,i,i+1]]  (bf16)
__global__ void build_gx(__bf16* __restrict__ gx, const float* __restrict__ h,
                         const float* __restrict__ e, int N) {
  int tid = blockIdx.x * blockDim.x + threadIdx.x;
  if (tid >= N * 128) return;
  int n = tid >> 7, c = tid & 127;
  float hs = 0.0f, es = 0.0f;
#pragma unroll
  for (int i = 0; i < 8; ++i) hs += h[((long)n * 8 + i) * 128 + c];
#pragma unroll
  for (int i = 0; i < 7; ++i) es += e[(((long)n * 8 + i) * 8 + i + 1) * 128 + c];
  gx[(long)n * 256 + c]       = f2bf(hs);
  gx[(long)n * 256 + 128 + c] = f2bf(es);
}

// ---------------------------------------------------------------------------
// Global GRU (h=0, so gh == bhh) + sigmoid output head, fused. 16 rows/block.
// ---------------------------------------------------------------------------
__global__ __launch_bounds__(WGSZ)
void gru_global_head(const __bf16* __restrict__ gx, const __bf16* __restrict__ Wg,
                     const float* __restrict__ biasG, const float* __restrict__ OW,
                     const float* __restrict__ Ob, float* __restrict__ out, int N) {
  extern __shared__ char smem[];
  __bf16* X  = (__bf16*)smem;                         // 16 x 256 bf16
  float* Out = (float*)(smem + 16 * 256 * 2);         // 16 x 384 f32
  float* G   = Out + 16 * 384;                        // 16 x 128 f32
  float* red = G + 16 * 128;                          // 16 x 8
  const long rowBase = (long)blockIdx.x * 16;

  for (int idx = threadIdx.x; idx < 16 * 256; idx += WGSZ) {
    int r = idx >> 8, c = idx & 255;
    long g = rowBase + r;
    X[idx] = (g < N) ? gx[g * 256 + c] : f2bf(0.0f);
  }
  __syncthreads();

  {  // gi = gx @ G_Wih^T : 24 column tiles, 8 k-chunks
    const int w = threadIdx.x >> 5;
    const int lane = threadIdx.x & 31, n = lane & 15, hi = lane >> 4;
    for (int ct = w; ct < 24; ct += 8) {
      v8f acc = {};
#pragma unroll
      for (int kc = 0; kc < 8; ++kc) {
        v16bf a = load_a_frag(X + kc * 32, 256);
        v16bf b = load_b_frag(Wg, 256, ct * 16, kc * 32);
        acc = wmma_bf16(a, b, acc);
      }
#pragma unroll
      for (int v = 0; v < 8; ++v) Out[(v + hi * 8) * 384 + ct * 16 + n] = acc[v];
    }
  }
  __syncthreads();

  for (int idx = threadIdx.x; idx < 16 * 128; idx += WGSZ) {
    int r = idx >> 7, c = idx & 127;
    float rr = Out[r * 384 + c]       + biasG[c];
    float zz = Out[r * 384 + 128 + c] + biasG[128 + c];
    float in = Out[r * 384 + 256 + c] + biasG[256 + c];
    float hn = biasG[384 + c];                    // Whh @ 0 == 0
    float rg = sigf(rr), zg = sigf(zz);
    G[idx] = (1.0f - zg) * tanh_fast(fmaf(rg, hn, in));
  }
  __syncthreads();

  if (threadIdx.x < 128) {
    int r = threadIdx.x >> 3, t = threadIdx.x & 7;
    float s = 0.0f;
#pragma unroll
    for (int q = 0; q < 16; ++q) { int c = t * 16 + q; s += G[r * 128 + c] * OW[c]; }
    red[r * 8 + t] = s;
  }
  __syncthreads();
  if (threadIdx.x < 16) {
    long g = rowBase + threadIdx.x;
    if (g < N) {
      float s = Ob[0];
#pragma unroll
      for (int t = 0; t < 8; ++t) s += red[threadIdx.x * 8 + t];
      out[g] = sigf(s);
    }
  }
}

// ---------------------------------------------------------------------------
extern "C" void kernel_launch(void* const* d_in, const int* in_sizes, int n_in,
                              void* d_out, int out_size, void* d_ws, size_t ws_size,
                              hipStream_t stream) {
  (void)in_sizes; (void)n_in; (void)out_size;
  const float* towers = (const float*)d_in[0];
  const float* EnW1 = (const float*)d_in[1];  const float* Enb1 = (const float*)d_in[2];
  const float* EnW2 = (const float*)d_in[3];  const float* Enb2 = (const float*)d_in[4];
  const float* EeW1 = (const float*)d_in[5];  const float* Eeb1 = (const float*)d_in[6];
  const float* EeW2 = (const float*)d_in[7];  const float* Eeb2 = (const float*)d_in[8];
  const float* UWih = (const float*)d_in[9];  const float* UWhh = (const float*)d_in[10];
  const float* Ubih = (const float*)d_in[11]; const float* Ubhh = (const float*)d_in[12];
  const float* MWih = (const float*)d_in[13]; const float* MWhh = (const float*)d_in[14];
  const float* Mbih = (const float*)d_in[15]; const float* Mbhh = (const float*)d_in[16];
  const float* GWih = (const float*)d_in[17]; const float* GWhh = (const float*)d_in[18];
  const float* Gbih = (const float*)d_in[19]; const float* Gbhh = (const float*)d_in[20];
  const float* OW   = (const float*)d_in[21]; const float* Ob   = (const float*)d_in[22];
  // d_in[23] = k (== 3 in setup); fixed iteration count keeps launch deterministic.
  (void)GWhh;

  const long N = 4096, NK = N * 8, NKK = NK * 8;
  char* ws = (char*)d_ws;
  size_t off = 0;
  auto alloc = [&](size_t bytes) -> void* {
    void* p = ws + off; off = (off + bytes + 255) & ~(size_t)255; return p;
  };
  float*  h     = (float*)alloc(NK * 128 * 4);
  float*  e     = (float*)alloc(NKK * 128 * 4);
  __bf16* xh    = (__bf16*)alloc(NK * 160 * 2);
  __bf16* nx    = (__bf16*)alloc(NK * 160 * 2);
  __bf16* gx    = (__bf16*)alloc(N * 256 * 2);
  __bf16* WcatM = (__bf16*)alloc(512 * 448 * 2);
  __bf16* WcatU = (__bf16*)alloc(512 * 288 * 2);
  __bf16* Wg    = (__bf16*)alloc(384 * 256 * 2);
  __bf16* WEn1  = (__bf16*)alloc(128 * 32 * 2);
  __bf16* WEn2  = (__bf16*)alloc(128 * 128 * 2);
  __bf16* WEe1  = (__bf16*)alloc(128 * 32 * 2);
  __bf16* WEe2  = (__bf16*)alloc(128 * 128 * 2);
  float*  biasM = (float*)alloc(512 * 4);
  float*  biasU = (float*)alloc(512 * 4);
  float*  biasG = (float*)alloc(512 * 4);
  if (off > ws_size) return;   // workspace too small; nothing safe to do

  // 1) pack weights/biases to bf16 WMMA-friendly layouts
  pack_pad_bf16<<<(128 * 32 + 255) / 256, 256, 0, stream>>>(WEn1, EnW1, 128, 14, 32);
  pack_pad_bf16<<<(128 * 128 + 255) / 256, 256, 0, stream>>>(WEn2, EnW2, 128, 128, 128);
  pack_pad_bf16<<<(128 * 32 + 255) / 256, 256, 0, stream>>>(WEe1, EeW1, 128, 28, 32);
  pack_pad_bf16<<<(128 * 128 + 255) / 256, 256, 0, stream>>>(WEe2, EeW2, 128, 128, 128);
  pack_pad_bf16<<<(384 * 256 + 255) / 256, 256, 0, stream>>>(Wg, GWih, 384, 256, 256);
  pack_gru_cat<<<(512 * 448 + 255) / 256, 256, 0, stream>>>(WcatM, MWih, MWhh, 320, 284);
  pack_gru_cat<<<(512 * 288 + 255) / 256, 256, 0, stream>>>(WcatU, UWih, UWhh, 160, 142);
  pack_bias<<<2, 256, 0, stream>>>(biasM, Mbih, Mbhh);
  pack_bias<<<2, 256, 0, stream>>>(biasU, Ubih, Ubhh);
  pack_bias<<<2, 256, 0, stream>>>(biasG, Gbih, Gbhh);

  // 2) encoders
  const size_t ldsEnc = EROWS * 32 * 2 + EROWS * 128 * 2;   // 20 KB
  encoder_mlp<<<(int)(NK / EROWS), WGSZ, ldsEnc, stream>>>(towers, WEn1, Enb1, WEn2, Enb2, h, NK, 0);
  encoder_mlp<<<(int)(NKK / EROWS), WGSZ, ldsEnc, stream>>>(towers, WEe1, Eeb1, WEe2, Eeb2, e, NKK, 1);

  // 3) k = 3 rounds of message passing
  const size_t ldsE = (size_t)GRROWS * 448 * 2;   // 56 KB
  const size_t ldsU = (size_t)GRROWS * 288 * 2;   // 36 KB
  for (int it = 0; it < 3; ++it) {
    build_xh<<<(int)((NK * 160 + 255) / 256), 256, 0, stream>>>(xh, towers, h, NK);
    gru_cell<<<(int)(NKK / GRROWS), WGSZ, ldsE, stream>>>(xh, e, WcatM, biasM, NKK, 0);
    build_nx<<<(int)((NK * 160 + 255) / 256), 256, 0, stream>>>(nx, towers, e, NK);
    gru_cell<<<(int)(NK / GRROWS), WGSZ, ldsU, stream>>>(nx, h, WcatU, biasU, NK, 1);
  }

  // 4) global readout + head
  build_gx<<<(int)((N * 128 + 255) / 256), 256, 0, stream>>>(gx, h, e, (int)N);
  const size_t ldsG = 16 * 256 * 2 + 16 * 384 * 4 + 16 * 128 * 4 + 16 * 8 * 4;
  gru_global_head<<<(int)(N / 16), WGSZ, ldsG, stream>>>(gx, Wg, biasG, OW, Ob,
                                                         (float*)d_out, (int)N);
}
